// DrugDrugInteractionNetwork_5231270167193
// MI455X (gfx1250) — matrix operand order repaired and checked by reading, compiled, not run on hardware
//
#include <hip/hip_runtime.h>

#define BGRAPHS 512
#define A_ATOMS 24
#define DEGE    4
#define EPER    (A_ATOMS*DEGE)      // 96
#define NATOMS  (BGRAPHS*A_ATOMS)   // 12288
#define DH      128
#define DFEAT   32
#define DRO     256
#define NLBL    12
#define NSTEPS  3
#define NBONDT  24

typedef __attribute__((ext_vector_type(16))) __bf16 v16bf;
typedef __attribute__((ext_vector_type(8)))  __bf16 v8bf;
typedef __attribute__((ext_vector_type(8)))  float  v8f;
typedef __attribute__((ext_vector_type(4)))  int    v4i;

__device__ __forceinline__ float leaky01(float x){ return x > 0.f ? x : 0.01f*x; }

#if defined(__HIP_DEVICE_COMPILE__) && defined(__gfx1250__) && \
    __has_builtin(__builtin_amdgcn_global_load_async_to_lds_b128) && \
    __has_builtin(__builtin_amdgcn_s_wait_asynccnt)
#define USE_ASYNC_LDS 1
#endif

// ---------------------------------------------------------------------------
// bf16 WMMA GEMM: C[M x Nn] = act(A[M x K] @ W[K x Nn] + bias)
// WT is the weight stored TRANSPOSED (Nn x K), so both the per-K-step LDS
// slab ([n][32] layout) and every lane's B fragment (16 contiguous bf16)
// are contiguous. Slab staging uses async global->LDS (ASYNCcnt) when the
// builtin exists. One block (8 waves) owns one 16-row M strip; all 8 waves
// share the LDS slab. A and B use contiguous K order (k = 16*half + e).
// Requires: Nn in {128, 256}, K multiple of 32, 16B-aligned buffers.
// ---------------------------------------------------------------------------
__global__ __launch_bounds__(256)
void wmma_gemm_bf16(const __bf16* __restrict__ A, const __bf16* __restrict__ WT,
                    const float* __restrict__ bias,
                    float* __restrict__ Cf, __bf16* __restrict__ Cb,
                    int M, int K, int Nn, int act)
{
    __shared__ __bf16 sBT[256 * 32];          // [n][k] slab, 16 KB (Nn<=256)
    int lane = threadIdx.x & 31;
    int wave = threadIdx.x >> 5;
    int tilesN = Nn >> 4;                     // 8 or 16
    int tm = blockIdx.x;                      // M strip index

    v8f acc0 = {0.f,0.f,0.f,0.f,0.f,0.f,0.f,0.f};
    v8f acc1 = {0.f,0.f,0.f,0.f,0.f,0.f,0.f,0.f};

    int arow  = tm*16 + (lane & 15);
    int aHalf = (lane >> 4) ? 16 : 0;         // A K-offset per lane half
    int bHalf = (lane >> 4) ? 16 : 0;         // B K-offset per lane half
    int bcol0 = wave*16 + (lane & 15);        // wave's first N tile column
    int bcol1 = (wave + 8)*16 + (lane & 15);  // wave's second N tile (Nn==256)

    for (int k0 = 0; k0 < K; k0 += 32) {
        // ---- stage K-slab of transposed weights: sBT[n][0..31] = WT[n][k0..k0+31]
        int chunks = Nn * 4;                  // 4 x 16B chunks per n row
#ifdef USE_ASYNC_LDS
        for (int i = threadIdx.x; i < chunks; i += 256) {
            int n = i >> 2, part = (i & 3) * 8;
            __builtin_amdgcn_global_load_async_to_lds_b128(
                (v4i __attribute__((address_space(1)))*)(WT + (size_t)n * K + k0 + part),
                (v4i __attribute__((address_space(3)))*)(sBT + n * 32 + part),
                0, 0);
        }
        __builtin_amdgcn_s_wait_asynccnt(0);
#else
        for (int i = threadIdx.x; i < chunks; i += 256) {
            int n = i >> 2, part = (i & 3) * 8;
            *(uint4*)(sBT + n * 32 + part) =
                *(const uint4*)(WT + (size_t)n * K + k0 + part);
        }
#endif
        __syncthreads();

        // ---- A fragment: two contiguous 16B vector loads per lane ----
        v16bf a;
        if (arow < M) {
            const __bf16* ap = A + (size_t)arow * K + k0 + aHalf;
            if (k0 + 32 < K) __builtin_prefetch(ap + 32, 0, 1);
            v8bf alo = *(const v8bf*)(ap);       // K = k0+aHalf+0..7
            v8bf ahi = *(const v8bf*)(ap + 8);   // K = k0+aHalf+8..15
#pragma unroll
            for (int e = 0; e < 8; ++e) { a[e] = alo[e]; a[8 + e] = ahi[e]; }
        } else {
#pragma unroll
            for (int e = 0; e < 16; ++e) a[e] = (__bf16)0.0f;
        }

        // ---- B fragments: one contiguous 32B LDS read per tile ----
        {
            v16bf b0 = *(const v16bf*)(sBT + bcol0 * 32 + bHalf);
            acc0 = __builtin_amdgcn_wmma_f32_16x16x32_bf16(false, a, false, b0,
                                                           (short)0, acc0, false, false);
        }
        if (tilesN > 8) {
            v16bf b1 = *(const v16bf*)(sBT + bcol1 * 32 + bHalf);
            acc1 = __builtin_amdgcn_wmma_f32_16x16x32_bf16(false, a, false, b1,
                                                           (short)0, acc1, false, false);
        }
        __syncthreads();   // protect sBT before next slab
    }

    // ---- store ----
#pragma unroll
    for (int g = 0; g < 8; ++g) {
        int row = tm*16 + g + ((lane >> 4) ? 8 : 0);
        if (row < M) {
            float v = acc0[g] + (bias ? bias[bcol0] : 0.f);
            if (act == 1)      v = leaky01(v);
            else if (act == 2) v = v > 0.f ? v : 0.f;
            if (Cf) Cf[(size_t)row * Nn + bcol0] = v;
            if (Cb) Cb[(size_t)row * Nn + bcol0] = (__bf16)v;
            if (tilesN > 8) {
                float w = acc1[g] + (bias ? bias[bcol1] : 0.f);
                if (act == 1)      w = leaky01(w);
                else if (act == 2) w = w > 0.f ? w : 0.f;
                if (Cf) Cf[(size_t)row * Nn + bcol1] = w;
                if (Cb) Cb[(size_t)row * Nn + bcol1] = (__bf16)w;
            }
        }
    }
}

// ---------------------------------------------------------------------------
// GraphPairNN: h = relu(ind @ W1 + b1), scores = h @ W2 + b2, preds = argmax
// ---------------------------------------------------------------------------
__global__ void pair_h_kernel(const float* __restrict__ W1, const float* __restrict__ b1,
                              float* __restrict__ h)
{
    int j = threadIdx.x;                 // 128 threads
    float s = b1[j];
    for (int i = 0; i < BGRAPHS; ++i) s += (float)i * W1[(size_t)i * 128 + j];
    h[j] = fmaxf(s, 0.f);
}

__global__ void pair_argmax_kernel(const float* __restrict__ h,
                                   const float* __restrict__ W2, const float* __restrict__ b2,
                                   int* __restrict__ preds, float* __restrict__ preds_f)
{
    __shared__ float sh[128];
    __shared__ float bv[128];
    __shared__ int   bi[128];
    int r = blockIdx.x, t = threadIdx.x;  // 128 threads
    sh[t] = h[t];
    __syncthreads();
    float best = -3.0e38f; int besti = 0x7fffffff;
    for (int c = t; c < BGRAPHS; c += 128) {
        size_t col = (size_t)r * BGRAPHS + c;
        float s = b2[col];
        for (int j = 0; j < 128; ++j) s += sh[j] * W2[(size_t)j * (BGRAPHS*BGRAPHS) + col];
        if (s > best) { best = s; besti = c; }
    }
    bv[t] = best; bi[t] = besti;
    __syncthreads();
    for (int stride = 64; stride > 0; stride >>= 1) {
        if (t < stride) {
            if (bv[t+stride] > bv[t] ||
                (bv[t+stride] == bv[t] && bi[t+stride] < bi[t])) {
                bv[t] = bv[t+stride]; bi[t] = bi[t+stride];
            }
        }
        __syncthreads();
    }
    if (t == 0) { preds[r] = bi[0]; preds_f[r] = (float)bi[0]; }
}

// ---------------------------------------------------------------------------
// Build bf16 A-matrix for node init: concat(atom_emb[type], feat), with
// optional whole-graph permutation via preds (graph 2).
// ---------------------------------------------------------------------------
__global__ void build_node_A(const int* __restrict__ atom_type, const float* __restrict__ atom_feat,
                             const float* __restrict__ atom_emb, const int* __restrict__ preds,
                             __bf16* __restrict__ Abuf)
{
    int idx = blockIdx.x * blockDim.x + threadIdx.x;  // i*160 + k
    if (idx >= NATOMS * (DH + DFEAT)) return;
    int i = idx / (DH + DFEAT), k = idx % (DH + DFEAT);
    int b = i / A_ATOMS, a = i % A_ATOMS;
    int src = preds ? preds[b] * A_ATOMS + a : i;
    float v;
    if (k < DH) v = atom_emb[(size_t)atom_type[src] * DH + k];
    else        v = atom_feat[(size_t)src * DFEAT + (k - DH)];
    Abuf[idx] = (__bf16)v;
}

__global__ void f2bf_kernel(const float* __restrict__ in, __bf16* __restrict__ out, int n)
{
    int i = blockIdx.x * blockDim.x + threadIdx.x;
    if (i < n) out[i] = (__bf16)in[i];
}

// convert f32 (K x Nn) -> bf16 transposed (Nn x K)
__global__ void f2bfT_kernel(const float* __restrict__ in, __bf16* __restrict__ out,
                             int K, int Nn)
{
    int idx = blockIdx.x * blockDim.x + threadIdx.x;
    if (idx >= K * Nn) return;
    int k = idx / Nn, n = idx % Nn;
    out[(size_t)n * K + k] = (__bf16)in[idx];
}

// ---------------------------------------------------------------------------
// Edge aggregation: im[i] = sum_{e: dst==i} leaky(T[src_e] + Ebt[bond_type_e])
// Edges are grouped DEGE per destination atom (ring-graph construction).
// preds != null applies the graph-2 block permutation to bond types / sources.
// ---------------------------------------------------------------------------
__global__ void edge_aggregate(const float* __restrict__ T, const float* __restrict__ Ebt,
                               const int* __restrict__ bond_type, const int* __restrict__ inn_idx,
                               const int* __restrict__ preds, float* __restrict__ im)
{
    int idx = blockIdx.x * blockDim.x + threadIdx.x;  // atom*128 + f
    if (idx >= NATOMS * DH) return;
    int atom = idx >> 7, f = idx & 127;
    int b = atom / A_ATOMS, a = atom % A_ATOMS;
    int srcg = preds ? preds[b] : b;
    float s = 0.f;
#pragma unroll
    for (int d = 0; d < DEGE; ++d) {
        int e_src = srcg * EPER + a * DEGE + d;       // edge pulled from source graph
        int bt = bond_type[e_src];
        int j_local = inn_idx[e_src] - srcg * A_ATOMS;
        int src_atom = b * A_ATOMS + j_local;
        s += leaky01(T[(size_t)src_atom * DH + f] + Ebt[(size_t)bt * DH + f]);
    }
    im[idx] = s;
}

// ---------------------------------------------------------------------------
// Cross-graph attention, one block per graph pair: 24x24 attention both ways,
// done on zero-padded 32x32 / 32x128 tiles entirely in LDS with WMMA.
// ---------------------------------------------------------------------------
__global__ __launch_bounds__(128)
void cross_attn(const float* __restrict__ K1f, const float* __restrict__ V1f,
                const float* __restrict__ K2f, const float* __restrict__ V2f,
                float* __restrict__ om1, float* __restrict__ om2)
{
    __shared__ __bf16 sK1[32*DH], sK2[32*DH], sV1[32*DH], sV2[32*DH];
    __shared__ float  sS[32*32];
    __shared__ __bf16 sP[32*32];
    int b = blockIdx.x;
    int tid = threadIdx.x;                // 128 threads = 4 waves
    int wave = tid >> 5, lane = tid & 31;

    for (int i = tid; i < 32*DH; i += 128) {
        int r = i >> 7, c = i & 127;
        float k1 = 0.f, k2 = 0.f, v1 = 0.f, v2 = 0.f;
        if (r < A_ATOMS) {
            size_t g = (size_t)(b * A_ATOMS + r) * DH + c;
            k1 = K1f[g]; k2 = K2f[g]; v1 = V1f[g]; v2 = V2f[g];
        }
        sK1[i] = (__bf16)k1; sK2[i] = (__bf16)k2;
        sV1[i] = (__bf16)v1; sV2[i] = (__bf16)v2;
    }
    __syncthreads();

    const float invTemp = 0.0883883476f;  // 1/sqrt(128)
    for (int dir = 0; dir < 2; ++dir) {
        const __bf16* Ka = dir ? sK2 : sK1;
        const __bf16* Kb = dir ? sK1 : sK2;
        const __bf16* Vb = dir ? sV1 : sV2;
        float* om = dir ? om2 : om1;

        // S = Ka @ Kb^T (32x32): 4 waves, one 16x16 tile each
        {
            int tm = wave >> 1, tn = wave & 1;
            v8f acc = {0.f,0.f,0.f,0.f,0.f,0.f,0.f,0.f};
            int arow = tm*16 + (lane & 15);
            int bcol = tn*16 + (lane & 15);
            int aHalf = (lane >> 4) ? 16 : 0;
            int bHalf = (lane >> 4) ? 16 : 0;
            for (int k0 = 0; k0 < DH; k0 += 32) {
                v16bf a, bb;
#pragma unroll
                for (int e = 0; e < 8; ++e) a[e]     = Ka[arow*DH + k0 + aHalf + e];
#pragma unroll
                for (int e = 0; e < 8; ++e) a[8 + e] = Ka[arow*DH + k0 + aHalf + 8 + e];
#pragma unroll
                for (int e = 0; e < 16; ++e) bb[e]   = Kb[bcol*DH + k0 + bHalf + e]; // B[k][n]=Kb[n][k]
                acc = __builtin_amdgcn_wmma_f32_16x16x32_bf16(false, a, false, bb,
                                                              (short)0, acc, false, false);
            }
#pragma unroll
            for (int g = 0; g < 8; ++g) {
                int row = tm*16 + g + ((lane >> 4) ? 8 : 0);
                sS[row*32 + bcol] = acc[g] * invTemp;
            }
        }
        __syncthreads();

        // row softmax over valid 24 cols; write zero-padded bf16 P
        if (tid < 32) {
            int r = tid;
            if (r < A_ATOMS) {
                float mx = -3.0e38f;
                for (int c = 0; c < A_ATOMS; ++c) mx = fmaxf(mx, sS[r*32 + c]);
                float ev[A_ATOMS]; float den = 0.f;
                for (int c = 0; c < A_ATOMS; ++c) { float e = __expf(sS[r*32 + c] - mx); ev[c] = e; den += e; }
                float inv = 1.f / (den + 1e-8f);
                for (int c = 0; c < 32; ++c)
                    sP[r*32 + c] = (__bf16)((c < A_ATOMS) ? ev[c] * inv : 0.f);
            } else {
                for (int c = 0; c < 32; ++c) sP[r*32 + c] = (__bf16)0.f;
            }
        }
        __syncthreads();

        // om = P @ Vb : (32x32)@(32x128) -> 2x8 tiles, 4 waves x 4 tiles, K=32
        for (int t = wave; t < 16; t += 4) {
            int tm = t >> 3, tn = t & 7;
            v8f acc = {0.f,0.f,0.f,0.f,0.f,0.f,0.f,0.f};
            v16bf a, bb;
            int arow = tm*16 + (lane & 15);
            int aHalf = (lane >> 4) ? 16 : 0;
            int bcol = tn*16 + (lane & 15);
            int bHalf = (lane >> 4) ? 16 : 0;
#pragma unroll
            for (int e = 0; e < 8; ++e) a[e]     = sP[arow*32 + aHalf + e];
#pragma unroll
            for (int e = 0; e < 8; ++e) a[8 + e] = sP[arow*32 + aHalf + 8 + e];
#pragma unroll
            for (int e = 0; e < 16; ++e) bb[e]   = Vb[(bHalf + e)*DH + bcol];
            acc = __builtin_amdgcn_wmma_f32_16x16x32_bf16(false, a, false, bb,
                                                          (short)0, acc, false, false);
#pragma unroll
            for (int g = 0; g < 8; ++g) {
                int row = tm*16 + g + ((lane >> 4) ? 8 : 0);
                if (row < A_ATOMS)
                    om[(size_t)(b*A_ATOMS + row)*DH + tn*16 + (lane & 15)] = acc[g];
            }
        }
        __syncthreads();
    }
}

// ---------------------------------------------------------------------------
// residual + LayerNorm, one block (128 threads) per atom row; wave32 shuffles
// ---------------------------------------------------------------------------
__global__ __launch_bounds__(128)
void combine_ln(const float* __restrict__ node, const float* __restrict__ im,
                const float* __restrict__ om,
                const float* __restrict__ ls, const float* __restrict__ lb,
                float* __restrict__ outf, __bf16* __restrict__ outb)
{
    __shared__ float red[4];
    int row = blockIdx.x, f = threadIdx.x;
    size_t o = (size_t)row * DH + f;
    float x = node[o] + im[o] + om[o];

    float s = x;
#pragma unroll
    for (int d = 16; d > 0; d >>= 1) s += __shfl_down(s, d, 32);
    if ((f & 31) == 0) red[f >> 5] = s;
    __syncthreads();
    float mean = (red[0] + red[1] + red[2] + red[3]) * (1.f / DH);
    __syncthreads();

    float dd = x - mean;
    float v = dd * dd;
#pragma unroll
    for (int d = 16; d > 0; d >>= 1) v += __shfl_down(v, d, 32);
    if ((f & 31) == 0) red[f >> 5] = v;
    __syncthreads();
    float var = (red[0] + red[1] + red[2] + red[3]) * (1.f / DH);

    float y = dd * rsqrtf(var + 1e-5f) * ls[f] + lb[f];
    outf[o] = y;
    outb[o] = (__bf16)y;
}

__global__ void pool_kernel(const float* __restrict__ node, float* __restrict__ gf,
                            __bf16* __restrict__ gb)
{
    int idx = blockIdx.x * blockDim.x + threadIdx.x;  // b*128+f
    if (idx >= BGRAPHS * DH) return;
    int b = idx >> 7, f = idx & 127;
    float s = 0.f;
    for (int a = 0; a < A_ATOMS; ++a) s += node[(size_t)(b*A_ATOMS + a)*DH + f];
    gf[idx] = s; gb[idx] = (__bf16)s;
}

__global__ void label_kernel(const float* __restrict__ g, const float* __restrict__ lW,
                             const float* __restrict__ lb, float* __restrict__ out)
{
    int idx = blockIdx.x * blockDim.x + threadIdx.x;  // b*12+l
    if (idx >= BGRAPHS * NLBL) return;
    int b = idx / NLBL, l = idx % NLBL;
    float s = lb[l];
    for (int j = 0; j < DRO; ++j) s += g[(size_t)b*DRO + j] * lW[(size_t)j*NLBL + l];
    out[idx] = s;
}

// ---------------------------------------------------------------------------
extern "C" void kernel_launch(void* const* d_in, const int* in_sizes, int n_in,
                              void* d_out, int out_size, void* d_ws, size_t ws_size,
                              hipStream_t stream)
{
    (void)in_sizes; (void)n_in; (void)out_size; (void)ws_size;
    const int*   atom_type1  = (const int*)  d_in[1];
    const float* atom_feat1  = (const float*)d_in[2];
    const int*   bond_type1  = (const int*)  d_in[3];
    const int*   inn_idx_j1  = (const int*)  d_in[5];
    const int*   atom_type2  = (const int*)  d_in[7];
    const float* atom_feat2  = (const float*)d_in[8];
    const int*   bond_type2  = (const int*)  d_in[9];
    const int*   inn_idx_j2  = (const int*)  d_in[11];
    const float* atom_emb    = (const float*)d_in[12];
    const float* bond_emb    = (const float*)d_in[13];
    const float* atom_proj_W = (const float*)d_in[14];
    const float* atom_proj_b = (const float*)d_in[15];
    const float* mp_node_W   = (const float*)d_in[16];
    const float* mp_edge_W   = (const float*)d_in[17];
    const float* mp_edge_b   = (const float*)d_in[18];
    const float* coat_key_W  = (const float*)d_in[19];
    const float* coat_val_W  = (const float*)d_in[20];
    const float* ln_scale    = (const float*)d_in[21];
    const float* ln_bias     = (const float*)d_in[22];
    const float* readout_W   = (const float*)d_in[23];
    const float* readout_b   = (const float*)d_in[24];
    const float* lbl_W       = (const float*)d_in[25];
    const float* lbl_b       = (const float*)d_in[26];
    const float* pair_W1     = (const float*)d_in[27];
    const float* pair_b1     = (const float*)d_in[28];
    const float* pair_W2     = (const float*)d_in[29];
    const float* pair_b2     = (const float*)d_in[30];

    // ---- workspace carve-out ----
    char* wp = (char*)d_ws;
    auto alloc = [&](size_t n) -> void* {
        void* r = (void*)wp; wp += (n + 255) & ~(size_t)255; return r;
    };
    const size_t NM = (size_t)NATOMS * DH;
    int*    preds  = (int*)   alloc(BGRAPHS * 4);
    float*  hbuf   = (float*) alloc(DH * 4);
    float*  node1f = (float*) alloc(NM * 4);
    float*  node2f = (float*) alloc(NM * 4);
    __bf16* node1b = (__bf16*)alloc(NM * 2);
    __bf16* node2b = (__bf16*)alloc(NM * 2);
    __bf16* Abuf   = (__bf16*)alloc((size_t)NATOMS * (DH + DFEAT) * 2);
    __bf16* WprojT = (__bf16*)alloc((size_t)(DH + DFEAT) * DH * 2);   // 128 x 160
    __bf16* WnodeT = (__bf16*)alloc((size_t)NSTEPS * DH * DH * 2);    // per step 128 x 128
    __bf16* WedgeT = (__bf16*)alloc((size_t)NSTEPS * DH * DH * 2);
    __bf16* WkeyT  = (__bf16*)alloc((size_t)NSTEPS * DH * DH * 2);
    __bf16* WvalT  = (__bf16*)alloc((size_t)NSTEPS * DH * DH * 2);
    __bf16* WroT   = (__bf16*)alloc((size_t)DH * DRO * 2);            // 256 x 128
    __bf16* bondAB = (__bf16*)alloc((size_t)NBONDT * DH * 2);
    float*  Ebtf   = (float*) alloc((size_t)NBONDT * DH * 4);
    float*  Tf     = (float*) alloc(NM * 4);
    float*  im1f   = (float*) alloc(NM * 4);
    float*  im2f   = (float*) alloc(NM * 4);
    float*  K1f    = (float*) alloc(NM * 4);
    float*  K2f    = (float*) alloc(NM * 4);
    float*  V1f    = (float*) alloc(NM * 4);
    float*  V2f    = (float*) alloc(NM * 4);
    float*  om1f   = (float*) alloc(NM * 4);
    float*  om2f   = (float*) alloc(NM * 4);
    float*  g1f    = (float*) alloc((size_t)BGRAPHS * DH * 4);
    float*  g2f    = (float*) alloc((size_t)BGRAPHS * DH * 4);
    __bf16* g1b    = (__bf16*)alloc((size_t)BGRAPHS * DH * 2);
    __bf16* g2b    = (__bf16*)alloc((size_t)BGRAPHS * DH * 2);
    float*  ro1f   = (float*) alloc((size_t)BGRAPHS * DRO * 4);
    float*  ro2f   = (float*) alloc((size_t)BGRAPHS * DRO * 4);

    auto cvt = [&](const float* src, __bf16* dst, int n) {
        f2bf_kernel<<<(n + 255) / 256, 256, 0, stream>>>(src, dst, n);
    };
    auto cvtT = [&](const float* src, __bf16* dst, int K, int Nn) {
        int n = K * Nn;
        f2bfT_kernel<<<(n + 255) / 256, 256, 0, stream>>>(src, dst, K, Nn);
    };
    auto gemm = [&](const __bf16* A, const __bf16* WT, const float* bias,
                    float* Cf, __bf16* Cb, int M, int K, int Nn, int act) {
        int tilesM = (M + 15) / 16;        // one block per 16-row strip
        wmma_gemm_bf16<<<tilesM, 256, 0, stream>>>(A, WT, bias, Cf, Cb, M, K, Nn, act);
    };

    // weight conversions: f32 (K x N) -> bf16 transposed (N x K), once
    cvtT(atom_proj_W, WprojT, DH + DFEAT, DH);
    for (int s = 0; s < NSTEPS; ++s) {
        cvtT(mp_node_W  + (size_t)s*DH*DH, WnodeT + (size_t)s*DH*DH, DH, DH);
        cvtT(mp_edge_W  + (size_t)s*DH*DH, WedgeT + (size_t)s*DH*DH, DH, DH);
        cvtT(coat_key_W + (size_t)s*DH*DH, WkeyT  + (size_t)s*DH*DH, DH, DH);
        cvtT(coat_val_W + (size_t)s*DH*DH, WvalT  + (size_t)s*DH*DH, DH, DH);
    }
    cvtT(readout_W, WroT, DH, DRO);
    cvt(bond_emb, bondAB, NBONDT * DH);    // A operand, stays row-major

    // pair matching -> preds (also written as float to output tail)
    float* out_f = (float*)d_out;
    pair_h_kernel<<<1, 128, 0, stream>>>(pair_W1, pair_b1, hbuf);
    pair_argmax_kernel<<<BGRAPHS, 128, 0, stream>>>(hbuf, pair_W2, pair_b2, preds,
                                                    out_f + 2 * BGRAPHS * NLBL);

    // node init (graph2 uses preds permutation)
    int nA = NATOMS * (DH + DFEAT);
    build_node_A<<<(nA + 255)/256, 256, 0, stream>>>(atom_type1, atom_feat1, atom_emb, nullptr, Abuf);
    gemm(Abuf, WprojT, atom_proj_b, node1f, node1b, NATOMS, DH + DFEAT, DH, 0);
    build_node_A<<<(nA + 255)/256, 256, 0, stream>>>(atom_type2, atom_feat2, atom_emb, preds, Abuf);
    gemm(Abuf, WprojT, atom_proj_b, node2f, node2b, NATOMS, DH + DFEAT, DH, 0);

    int aggBlocks = (int)((NM + 255) / 256);
    for (int s = 0; s < NSTEPS; ++s) {
        const __bf16* nW = WnodeT + (size_t)s * DH * DH;
        const __bf16* eW = WedgeT + (size_t)s * DH * DH;
        const __bf16* kW = WkeyT  + (size_t)s * DH * DH;
        const __bf16* vW = WvalT  + (size_t)s * DH * DH;
        const float*  eb = mp_edge_b + (size_t)s * DH;
        const float*  ls = ln_scale  + (size_t)s * DH;
        const float*  lb = ln_bias   + (size_t)s * DH;

        // per-bond-type edge transform: (24 x 128) @ (128 x 128) + eb
        gemm(bondAB, eW, eb, Ebtf, nullptr, NBONDT, DH, DH, 0);

        // graph 1 inner messages
        gemm(node1b, nW, nullptr, Tf, nullptr, NATOMS, DH, DH, 0);
        edge_aggregate<<<aggBlocks, 256, 0, stream>>>(Tf, Ebtf, bond_type1, inn_idx_j1,
                                                      nullptr, im1f);
        // graph 2 inner messages (permuted)
        gemm(node2b, nW, nullptr, Tf, nullptr, NATOMS, DH, DH, 0);
        edge_aggregate<<<aggBlocks, 256, 0, stream>>>(Tf, Ebtf, bond_type2, inn_idx_j2,
                                                      preds, im2f);
        // keys / values
        gemm(node1b, kW, nullptr, K1f, nullptr, NATOMS, DH, DH, 0);
        gemm(node1b, vW, nullptr, V1f, nullptr, NATOMS, DH, DH, 0);
        gemm(node2b, kW, nullptr, K2f, nullptr, NATOMS, DH, DH, 0);
        gemm(node2b, vW, nullptr, V2f, nullptr, NATOMS, DH, DH, 0);

        // cross-graph attention, one block per matched pair
        cross_attn<<<BGRAPHS, 128, 0, stream>>>(K1f, V1f, K2f, V2f, om1f, om2f);

        // residual + layernorm (in place; keeps bf16 mirror fresh)
        combine_ln<<<NATOMS, 128, 0, stream>>>(node1f, im1f, om1f, ls, lb, node1f, node1b);
        combine_ln<<<NATOMS, 128, 0, stream>>>(node2f, im2f, om2f, ls, lb, node2f, node2b);
    }

    // readout
    pool_kernel<<<(BGRAPHS*DH + 255)/256, 256, 0, stream>>>(node1f, g1f, g1b);
    pool_kernel<<<(BGRAPHS*DH + 255)/256, 256, 0, stream>>>(node2f, g2f, g2b);
    gemm(g1b, WroT, readout_b, ro1f, nullptr, BGRAPHS, DH, DRO, 1);
    gemm(g2b, WroT, readout_b, ro2f, nullptr, BGRAPHS, DH, DRO, 1);
    label_kernel<<<(BGRAPHS*NLBL + 255)/256, 256, 0, stream>>>(ro1f, lbl_W, lbl_b, out_f);
    label_kernel<<<(BGRAPHS*NLBL + 255)/256, 256, 0, stream>>>(ro2f, lbl_W, lbl_b,
                                                               out_f + BGRAPHS * NLBL);
}
